// GIN_46153718563030
// MI455X (gfx1250) — matrix operand order repaired and checked by reading, not compiled, not run on hardware
//
#include <hip/hip_runtime.h>

#define D 128
#define LDSW 136  // padded LDS row stride in halves (breaks 256B bank period)

typedef __attribute__((ext_vector_type(16))) _Float16 v16h;
typedef __attribute__((ext_vector_type(8)))  _Float16 v8h;
typedef __attribute__((ext_vector_type(4)))  _Float16 v4h;
typedef __attribute__((ext_vector_type(8)))  float    v8f;
typedef __attribute__((ext_vector_type(4)))  float    v4f;

// ---------------------------------------------------------------------------
// h = X @ W^T + b   (W is [128 out][128 in], f32)
// Dual-store: Y (h buffer, read by the edge scatter) and Y2 (agg buffer,
// pre-initialized to h so the scatter can accumulate on top: GIN eps=0).
// 8 waves/block, 16 rows/wave, f16 WMMA with f32 accumulate.
// ---------------------------------------------------------------------------
__global__ __launch_bounds__(256) void linear_wmma(const float* __restrict__ X,
                                                   const float* __restrict__ W,
                                                   const float* __restrict__ Bv,
                                                   float* __restrict__ Y,
                                                   float* __restrict__ Y2, int n) {
    __shared__ _Float16 Wh[D * LDSW];

    // Cooperative f32 -> f16 staging of the weight matrix into LDS.
    for (int i = threadIdx.x; i < D * D; i += 256) {
        int r = i >> 7, c = i & 127;
        Wh[r * LDSW + c] = (_Float16)W[i];
    }
    __syncthreads();

    const int wave = threadIdx.x >> 5;
    const int lane = threadIdx.x & 31;
    const int mrow = lane & 15;   // A: row owned by this lane; B/C/D: column
    const int hi   = lane >> 4;   // lane half selector
    const int rowBase = blockIdx.x * 128 + wave * 16;
    const int row   = rowBase + mrow;
    const int rsafe = row < n ? row : (n - 1);
    const bool fullTile = (rowBase + 16 <= n);   // wave-uniform
    const bool anyRow   = (rowBase < n);         // wave-uniform

    // --- A fragments: 16x32 f16 per chunk, ISA layout:
    //     lanes 0-15:  K = {0..7, 16..23};  lanes 16-31: K = {8..15, 24..31}
    v16h afrag[4];
    const float* xrow = X + (size_t)rsafe * D;
#pragma unroll
    for (int kc = 0; kc < 4; ++kc) {
        const v4f* p = (const v4f*)(xrow + kc * 32 + hi * 8);
        v4f q0 = p[0], q1 = p[1], q2 = p[4], q3 = p[5];
        v4h h0 = __builtin_convertvector(q0, v4h);
        v4h h1 = __builtin_convertvector(q1, v4h);
        v4h h2 = __builtin_convertvector(q2, v4h);
        v4h h3 = __builtin_convertvector(q3, v4h);
        v8h lo = __builtin_shufflevector(h0, h1, 0, 1, 2, 3, 4, 5, 6, 7);
        v8h hh = __builtin_shufflevector(h2, h3, 0, 1, 2, 3, 4, 5, 6, 7);
        afrag[kc] = __builtin_shufflevector(lo, hh, 0, 1, 2, 3, 4, 5, 6, 7,
                                            8, 9, 10, 11, 12, 13, 14, 15);
    }

    const int khalf = hi * 8;    // B: lanes 0-15 hold K={0..7,16..23}, lanes 16-31 K={8..15,24..31}
#pragma unroll
    for (int ct = 0; ct < 8; ++ct) {
        const int col = ct * 16 + mrow;          // output feature column
        const float bias = Bv[col];
        v8f acc;
#pragma unroll
        for (int r = 0; r < 8; ++r) acc[r] = bias;

#pragma unroll
        for (int kc = 0; kc < 4; ++kc) {
            const v8h* wp = (const v8h*)(&Wh[col * LDSW + kc * 32 + khalf]);
            v8h blo = wp[0], bhi = wp[2];
            v16h bfrag = __builtin_shufflevector(blo, bhi, 0, 1, 2, 3, 4, 5, 6, 7,
                                                 8, 9, 10, 11, 12, 13, 14, 15);
            acc = __builtin_amdgcn_wmma_f32_16x16x32_f16(
                false, afrag[kc], false, bfrag, (short)0, acc, false, false);
        }

        // D layout: VGPR r, lanes 0-15 -> M=r, lanes 16-31 -> M=r+8
        if (fullTile) {
#pragma unroll
            for (int r = 0; r < 8; ++r) {
                size_t off = (size_t)(rowBase + r + 8 * hi) * D + col;
                Y[off]  = acc[r];
                Y2[off] = acc[r];
            }
        } else if (anyRow) {
#pragma unroll
            for (int r = 0; r < 8; ++r) {
                int m = rowBase + r + 8 * hi;
                if (m < n) {
                    size_t off = (size_t)m * D + col;
                    Y[off]  = acc[r];
                    Y2[off] = acc[r];
                }
            }
        }
    }
}

// ---------------------------------------------------------------------------
// AGG[dst[e], :] += H[src[e], :]   one warp per edge, 4 floats per lane.
// ---------------------------------------------------------------------------
__global__ __launch_bounds__(256) void scatter_add_edges(const float* __restrict__ H,
                                                         float* __restrict__ AGG,
                                                         const int* __restrict__ src,
                                                         const int* __restrict__ dst,
                                                         int nE) {
    long long tid = (long long)blockIdx.x * 256 + threadIdx.x;
    int e = (int)(tid >> 5);
    if (e >= nE) return;
    int seg = (int)(tid & 31) * 4;
    int s = src[e], d = dst[e];
    v4f v = *(const v4f*)(H + (size_t)s * D + seg);
    float* ap = AGG + (size_t)d * D + seg;
    unsafeAtomicAdd(ap + 0, v[0]);
    unsafeAtomicAdd(ap + 1, v[1]);
    unsafeAtomicAdd(ap + 2, v[2]);
    unsafeAtomicAdd(ap + 3, v[3]);
}

// out = relu(a)   (LeakyReLU after ReLU is a no-op; agg already contains +h)
__global__ __launch_bounds__(256) void relu_kernel(const float* __restrict__ A,
                                                   float* __restrict__ Out,
                                                   long long n4) {
    long long i = (long long)blockIdx.x * 256 + threadIdx.x;
    if (i >= n4) return;
    v4f r = ((const v4f*)A)[i];
#pragma unroll
    for (int k = 0; k < 4; ++k) r[k] = r[k] > 0.f ? r[k] : 0.f;
    ((v4f*)Out)[i] = r;
}

// ---------------------------------------------------------------------------
// logits = EMB @ Wout^T + bout (3 classes), ypred = argmax (first max).
// ---------------------------------------------------------------------------
__global__ __launch_bounds__(256) void head_kernel(const float* __restrict__ EMB,
                                                   const float* __restrict__ Wout,
                                                   const float* __restrict__ bout,
                                                   float* __restrict__ logits,
                                                   float* __restrict__ ypred, int n) {
    __shared__ __align__(16) float Ws[3 * D];
    __shared__ float bs[3];
    for (int i = threadIdx.x; i < 3 * D; i += 256) Ws[i] = Wout[i];
    if (threadIdx.x < 3) bs[threadIdx.x] = bout[threadIdx.x];
    __syncthreads();

    int nid = blockIdx.x * 256 + threadIdx.x;
    if (nid >= n) return;

    const v4f* e = (const v4f*)(EMB + (size_t)nid * D);
    float acc0 = bs[0], acc1 = bs[1], acc2 = bs[2];
#pragma unroll 4
    for (int k4 = 0; k4 < D / 4; ++k4) {
        v4f v = e[k4];
        v4f w0 = ((const v4f*)&Ws[0 * D])[k4];
        v4f w1 = ((const v4f*)&Ws[1 * D])[k4];
        v4f w2 = ((const v4f*)&Ws[2 * D])[k4];
#pragma unroll
        for (int k = 0; k < 4; ++k) {
            acc0 += v[k] * w0[k];
            acc1 += v[k] * w1[k];
            acc2 += v[k] * w2[k];
        }
    }
    logits[(size_t)nid * 3 + 0] = acc0;
    logits[(size_t)nid * 3 + 1] = acc1;
    logits[(size_t)nid * 3 + 2] = acc2;

    float best = acc0; int bi = 0;               // first-occurrence argmax
    if (acc1 > best) { best = acc1; bi = 1; }
    if (acc2 > best) { best = acc2; bi = 2; }
    ypred[nid] = (float)bi;
}

__global__ __launch_bounds__(256) void gather_kernel(const float* __restrict__ logits,
                                                     const float* __restrict__ ypred,
                                                     const int* __restrict__ nodeIdx,
                                                     float* __restrict__ node_out,
                                                     float* __restrict__ y_nodepred, int m) {
    int i = blockIdx.x * 256 + threadIdx.x;
    if (i >= m) return;
    int ni = nodeIdx[i];
    node_out[(size_t)i * 3 + 0] = logits[(size_t)ni * 3 + 0];
    node_out[(size_t)i * 3 + 1] = logits[(size_t)ni * 3 + 1];
    node_out[(size_t)i * 3 + 2] = logits[(size_t)ni * 3 + 2];
    y_nodepred[i] = ypred[ni];
}

// ---------------------------------------------------------------------------
extern "C" void kernel_launch(void* const* d_in, const int* in_sizes, int n_in,
                              void* d_out, int out_size, void* d_ws, size_t ws_size,
                              hipStream_t stream) {
    const float* x    = (const float*)d_in[0];
    const int*   ei   = (const int*)d_in[1];
    // d_in[2] = node_label (unused by reference computation)
    const int*   nidx = (const int*)d_in[3];
    const float* W1 = (const float*)d_in[4];  const float* b1 = (const float*)d_in[5];
    const float* W2 = (const float*)d_in[6];  const float* b2 = (const float*)d_in[7];
    const float* W3 = (const float*)d_in[8];  const float* b3 = (const float*)d_in[9];
    const float* Wo = (const float*)d_in[10]; const float* bo = (const float*)d_in[11];

    const int n  = in_sizes[0] / D;
    const int nE = in_sizes[1] / 2;
    const int m  = in_sizes[3];
    const int* src = ei;
    const int* dst = ei + nE;

    float* buf0 = (float*)d_ws;                 // h / activations
    float* buf1 = buf0 + (size_t)n * D;         // agg (reused for logits at end)
    float* buf2 = buf1 + (size_t)n * D;

    float* out_embed = (float*)d_out;                        // [n,128]
    float* out_nout  = out_embed + (size_t)n * D;            // [m,3]
    float* out_ypred = out_nout + (size_t)m * 3;             // [n]
    float* out_ynp   = out_ypred + (size_t)n;                // [m]
    float* logits    = buf1;                                 // free after layer 3

    dim3 blk(256);
    const int gLin = (n + 127) / 128;
    const long long n4 = (long long)n * (D / 4);
    const int gElem = (int)((n4 + 255) / 256);
    const long long sth = (long long)nE * 32;
    const int gScat = (int)((sth + 255) / 256);

    // Layer 1: h -> buf0, agg(init=h) -> buf1; scatter adds; relu -> buf0
    linear_wmma<<<gLin, blk, 0, stream>>>(x, W1, b1, buf0, buf1, n);
    scatter_add_edges<<<gScat, blk, 0, stream>>>(buf0, buf1, src, dst, nE);
    relu_kernel<<<gElem, blk, 0, stream>>>(buf1, buf0, n4);

    // Layer 2: h -> buf2, agg(init=h) -> buf1; relu -> buf2
    linear_wmma<<<gLin, blk, 0, stream>>>(buf0, W2, b2, buf2, buf1, n);
    scatter_add_edges<<<gScat, blk, 0, stream>>>(buf2, buf1, src, dst, nE);
    relu_kernel<<<gElem, blk, 0, stream>>>(buf1, buf2, n4);

    // Layer 3: h -> buf0, agg(init=h) -> buf1; relu -> x_embed in d_out
    linear_wmma<<<gLin, blk, 0, stream>>>(buf2, W3, b3, buf0, buf1, n);
    scatter_add_edges<<<gScat, blk, 0, stream>>>(buf0, buf1, src, dst, nE);
    relu_kernel<<<gElem, blk, 0, stream>>>(buf1, out_embed, n4);

    // Head + gathers
    head_kernel<<<(n + 255) / 256, blk, 0, stream>>>(out_embed, Wo, bo, logits,
                                                     out_ypred, n);
    gather_kernel<<<(m + 255) / 256, blk, 0, stream>>>(logits, out_ypred, nidx,
                                                       out_nout, out_ynp, m);
}
